// ChunkwiseRetention_28217935135456
// MI455X (gfx1250) — matrix operand order repaired and checked by reading, not compiled, not run on hardware
//
#include <hip/hip_runtime.h>
#include <hip/hip_bf16.h>
#include <math.h>

// ---------------------------------------------------------------------------
// ChunkwiseRetention for MI455X (gfx1250, wave32, WMMA 16x16x32 bf16)
// B=8, T=1024, C=1024, NH=16, HD=64
// ---------------------------------------------------------------------------

#define B_ 8
#define T_ 1024
#define C_ 1024
#define NH_ 16
#define HD_ 64
#define M_ (B_ * T_)        // 8192 rows of the QKV GEMM
#define N3C_ (3 * C_)       // 3072

typedef __bf16 bf16_t;
typedef __attribute__((ext_vector_type(16))) __bf16 v16bf;
typedef __attribute__((ext_vector_type(8)))  __bf16 v8bf;
typedef __attribute__((ext_vector_type(8)))  float  v8f;

// ---------------------------------------------------------------------------
// WMMA fragment helpers (A/B in contiguous k order, C/D per ISA, wave32)
// ---------------------------------------------------------------------------

// A fragment (16x32 bf16, M x K): lane l holds row m = l%16.
// lanes 0-15:  K = [0..15]
// lanes 16-31: K = [16..31]
__device__ inline v16bf load_a_frag(const bf16_t* __restrict__ tile, int ldm) {
  const int lane = threadIdx.x & 31;
  const bf16_t* p = tile + (size_t)(lane & 15) * ldm + ((lane >> 4) << 4);
  v8bf lo = *(const v8bf*)p;
  v8bf hi = *(const v8bf*)(p + 8);
  v16bf a;
#pragma unroll
  for (int i = 0; i < 8; ++i) { a[i] = lo[i]; a[8 + i] = hi[i]; }
  return a;
}

// LDS flavor of the A-fragment load (for the S repack in the retention kernel)
__device__ inline v16bf load_a_frag_lds(const bf16_t* tile, int ldm) {
  const int lane = threadIdx.x & 31;
  const bf16_t* p = tile + (lane & 15) * ldm + ((lane >> 4) << 4);
  v8bf lo = *(const v8bf*)p;
  v8bf hi = *(const v8bf*)(p + 8);
  v16bf a;
#pragma unroll
  for (int i = 0; i < 8; ++i) { a[i] = lo[i]; a[8 + i] = hi[i]; }
  return a;
}

// B fragment (32x16 bf16, K x N), source stored N-major (one row per output
// column, K contiguous): lane l = column n = l%16, holds K chunk
// [(l/16)*16 .. +16) as 16 contiguous bf16 (one 32B load).
__device__ inline v16bf load_b_frag(const bf16_t* __restrict__ tileT, int ldn) {
  const int lane = threadIdx.x & 31;
  return *(const v16bf*)(tileT + (size_t)(lane & 15) * ldn + ((lane >> 4) << 4));
}

__device__ inline v8f wmma_bf16(v16bf a, v16bf b, v8f c) {
  // emits v_wmma_f32_16x16x32_bf16
  return __builtin_amdgcn_wmma_f32_16x16x32_bf16(
      /*neg_a=*/false, a, /*neg_b=*/false, b,
      /*c_mod=*/(short)0, c, /*reuse_a=*/false, /*reuse_b=*/false);
}

// C/D layout: VGPR r holds (m = r + (lane>=16 ? 8 : 0), n = lane%16)
__device__ inline int cd_row(int r) {
  return r + (((threadIdx.x & 31) >> 4) << 3);
}
__device__ inline int cd_col() { return threadIdx.x & 15; }

// ---------------------------------------------------------------------------
// 0) workspace zero-init (sums + kv accumulator; harness poisons ws with 0xAA)
// ---------------------------------------------------------------------------
__global__ void init_kernel(float* __restrict__ sums, float* __restrict__ kv_accum) {
  int i = blockIdx.x * blockDim.x + threadIdx.x;
  if (i < 2 * B_) sums[i] = 0.0f;
  if (i < NH_ * HD_ * HD_) kv_accum[i] = 0.0f;
}

// ---------------------------------------------------------------------------
// 1) casts / transposes into bf16 operand layouts
// ---------------------------------------------------------------------------
__global__ void cast_x_kernel(const float* __restrict__ x, bf16_t* __restrict__ xb) {
  size_t i = (size_t)blockIdx.x * blockDim.x + threadIdx.x;
  if (i < (size_t)M_ * C_) xb[i] = (bf16_t)x[i];
}

// W_qkv [C, 3C] row-major  ->  Wt [3C, C] (each output feature's weights
// contiguous along K, ready for the B-operand layout)
__global__ void cast_w_kernel(const float* __restrict__ W, bf16_t* __restrict__ Wt) {
  size_t i = (size_t)blockIdx.x * blockDim.x + threadIdx.x;
  if (i >= (size_t)N3C_ * C_) return;
  int n = (int)(i / C_);
  int k = (int)(i % C_);
  Wt[i] = (bf16_t)W[(size_t)k * N3C_ + n];
}

// past_kv [NH, HD, HD] (h,d,e) -> pkT [NH, HD, HD] (h,e,d), scaled by
// inner_decay = NH*HD = 1024 (folded into the cross-term operand)
__global__ void cast_pk_kernel(const float* __restrict__ pk, bf16_t* __restrict__ pkT) {
  int i = blockIdx.x * blockDim.x + threadIdx.x;
  if (i >= NH_ * HD_ * HD_) return;
  int h = i >> 12;
  int e = (i >> 6) & 63;
  int d = i & 63;
  pkT[i] = (bf16_t)(pk[(h << 12) + (d << 6) + e] * (float)(NH_ * HD_));
}

// ---------------------------------------------------------------------------
// 2) QKV GEMM: [8192 x 1024] x [1024 x 3072], wave tile = 32(M) x 64(N),
//    K step 32. Software-pipelined: A fragments double-buffered, the 4 B
//    fragments batch-loaded (one s_clause) at the top of each K step, and
//    8 named accumulators so the register allocator pins them (no v_mov /
//    WMMA->VALU hazard NOPs). Fused split-heads + bf16 cast + transposes.
// ---------------------------------------------------------------------------
__global__ __launch_bounds__(256) void qkv_gemm_kernel(
    const bf16_t* __restrict__ xb, const bf16_t* __restrict__ Wt,
    bf16_t* __restrict__ qb, bf16_t* __restrict__ kb,
    bf16_t* __restrict__ kTb, bf16_t* __restrict__ vTb) {
  const int lane = threadIdx.x & 31;
  const int wave = (int)((blockIdx.x * blockDim.x + threadIdx.x) >> 5);
  const int nT = wave % (N3C_ / 64);
  const int mT = wave / (N3C_ / 64);
  if (mT >= M_ / 32) return;
  const int mBase = mT * 32;
  const int nBase = nT * 64;
  const int kind = nBase >> 10;           // 0=q 1=k 2=v
  const int h = (nBase & 1023) >> 6;

  const bf16_t* pa0 = xb + (size_t)mBase * C_;
  const bf16_t* pa1 = xb + (size_t)(mBase + 16) * C_;
  const bf16_t* pb = Wt + (size_t)nBase * C_;

  v8f zero = {};
  v8f c00 = zero, c01 = zero, c02 = zero, c03 = zero;
  v8f c10 = zero, c11 = zero, c12 = zero, c13 = zero;

  // prologue: A fragments for k=0
  v16bf a0 = load_a_frag(pa0, C_);
  v16bf a1 = load_a_frag(pa1, C_);

  for (int kc = 0; kc < C_ / 32; ++kc) {
    const int k0 = kc * 32;
    // batch the 4 B-fragment loads (clause together)
    v16bf b0 = load_b_frag(pb + (size_t)0 * 16 * C_ + k0, C_);
    v16bf b1 = load_b_frag(pb + (size_t)1 * 16 * C_ + k0, C_);
    v16bf b2 = load_b_frag(pb + (size_t)2 * 16 * C_ + k0, C_);
    v16bf b3 = load_b_frag(pb + (size_t)3 * 16 * C_ + k0, C_);
    // double-buffer: next-K A fragments in flight under the WMMAs
    v16bf na0 = a0, na1 = a1;
    if (kc + 1 < C_ / 32) {
      na0 = load_a_frag(pa0 + k0 + 32, C_);
      na1 = load_a_frag(pa1 + k0 + 32, C_);
      __builtin_prefetch(pb + (size_t)(lane & 15) * C_ + k0 + 64, 0, 1);
    }
    c00 = wmma_bf16(a0, b0, c00);
    c10 = wmma_bf16(a1, b0, c10);
    c01 = wmma_bf16(a0, b1, c01);
    c11 = wmma_bf16(a1, b1, c11);
    c02 = wmma_bf16(a0, b2, c02);
    c12 = wmma_bf16(a1, b2, c12);
    c03 = wmma_bf16(a0, b3, c03);
    c13 = wmma_bf16(a1, b3, c13);
    a0 = na0;
    a1 = na1;
  }

  // epilogue: scatter bf16 into head-major layouts
  v8f accs[2][4] = {{c00, c01, c02, c03}, {c10, c11, c12, c13}};
#pragma unroll
  for (int rt = 0; rt < 2; ++rt) {
#pragma unroll
    for (int ct = 0; ct < 4; ++ct) {
      const int d = ct * 16 + cd_col();
#pragma unroll
      for (int rr = 0; rr < 8; ++rr) {
        const int m = mBase + rt * 16 + cd_row(rr);
        const int b = m >> 10;
        const int t = m & 1023;
        const size_t bh = (size_t)(b * NH_ + h);
        const bf16_t val = (bf16_t)accs[rt][ct][rr];
        if (kind == 0) {
          qb[(bh * T_ + t) * HD_ + d] = val;
        } else if (kind == 1) {
          kb[(bh * T_ + t) * HD_ + d] = val;
          kTb[(bh * HD_ + d) * T_ + t] = val;
        } else {
          vTb[(bh * HD_ + d) * T_ + t] = val;
        }
      }
    }
  }
}

// ---------------------------------------------------------------------------
// 3) Retention: per wave = (b, h, 16-row tile of T).
//    acc  = q @ pkT(*1024)                    (cross term)
//    acc += (decay ∘ (q @ k^T)) @ v           (triangular, K chunks of 32)
//    The decayed S tile is repacked C-layout -> A-layout through LDS.
//    Also accumulates GroupNorm sum / sumsq via shuffle + f32 atomics.
// ---------------------------------------------------------------------------
__global__ __launch_bounds__(256) void retention_kernel(
    const bf16_t* __restrict__ q, const bf16_t* __restrict__ k,
    const bf16_t* __restrict__ vT, const bf16_t* __restrict__ pkT,
    float* __restrict__ rbuf, float* __restrict__ sums) {
  __shared__ bf16_t lds[8][16 * 32];      // one 16x32 bf16 S tile per wave
  const int lane = threadIdx.x & 31;
  const int wib = threadIdx.x >> 5;
  const int wave = (int)(blockIdx.x * 8 + wib);   // 0 .. B*NH*64-1
  const int rt = wave & 63;
  const int bh = wave >> 6;
  const int h = bh & (NH_ - 1);
  const int b = bh >> 4;

  const float gamma = 1.0f - exp2f(-5.0f - (float)h);
  const float l2g = log2f(gamma);

  const bf16_t* qh = q + (size_t)bh * T_ * HD_;
  const bf16_t* kh = k + (size_t)bh * T_ * HD_;
  const bf16_t* vTh = vT + (size_t)bh * HD_ * T_;
  const bf16_t* pkh = pkT + (size_t)h * HD_ * HD_;

  const int rowBase = rt * 16;

  // q A-fragments for both K(d) chunks of HD=64
  v16bf aq0 = load_a_frag(qh + (size_t)rowBase * HD_ + 0, HD_);
  v16bf aq1 = load_a_frag(qh + (size_t)rowBase * HD_ + 32, HD_);

  v8f zero = {};
  v8f acc0 = zero, acc1 = zero, acc2 = zero, acc3 = zero;

  // cross term: q @ pkT (pkT already scaled by NH*HD); batch the 8 loads
  {
    v16bf p00 = load_b_frag(pkh + (size_t)0 * 16 * HD_ + 0, HD_);
    v16bf p01 = load_b_frag(pkh + (size_t)0 * 16 * HD_ + 32, HD_);
    v16bf p10 = load_b_frag(pkh + (size_t)1 * 16 * HD_ + 0, HD_);
    v16bf p11 = load_b_frag(pkh + (size_t)1 * 16 * HD_ + 32, HD_);
    v16bf p20 = load_b_frag(pkh + (size_t)2 * 16 * HD_ + 0, HD_);
    v16bf p21 = load_b_frag(pkh + (size_t)2 * 16 * HD_ + 32, HD_);
    v16bf p30 = load_b_frag(pkh + (size_t)3 * 16 * HD_ + 0, HD_);
    v16bf p31 = load_b_frag(pkh + (size_t)3 * 16 * HD_ + 32, HD_);
    acc0 = wmma_bf16(aq0, p00, acc0);
    acc0 = wmma_bf16(aq1, p01, acc0);
    acc1 = wmma_bf16(aq0, p10, acc1);
    acc1 = wmma_bf16(aq1, p11, acc1);
    acc2 = wmma_bf16(aq0, p20, acc2);
    acc2 = wmma_bf16(aq1, p21, acc2);
    acc3 = wmma_bf16(aq0, p30, acc3);
    acc3 = wmma_bf16(aq1, p31, acc3);
  }

  // triangular intra-chunk part, 32 s-columns per iteration
  const int nChunks = (rowBase + 15) / 32 + 1;
  bf16_t* tile = lds[wib];
  for (int jc = 0; jc < nChunks; ++jc) {
    const int s0 = jc * 32;
    // S = q @ k^T for this 16x32 column slab (4 batched B loads, 4 WMMA)
    v16bf bk00 = load_b_frag(kh + (size_t)(s0 + 0) * HD_ + 0, HD_);
    v16bf bk01 = load_b_frag(kh + (size_t)(s0 + 0) * HD_ + 32, HD_);
    v16bf bk10 = load_b_frag(kh + (size_t)(s0 + 16) * HD_ + 0, HD_);
    v16bf bk11 = load_b_frag(kh + (size_t)(s0 + 16) * HD_ + 32, HD_);
    v8f s0v = zero, s1v = zero;
    s0v = wmma_bf16(aq0, bk00, s0v);
    s0v = wmma_bf16(aq1, bk01, s0v);
    s1v = wmma_bf16(aq0, bk10, s1v);
    s1v = wmma_bf16(aq1, bk11, s1v);
    // decay mask + repack (C layout -> row-major 16x32 bf16 in LDS)
#pragma unroll
    for (int rr = 0; rr < 8; ++rr) {
      const int m = cd_row(rr);
      const int i = rowBase + m;
      const int j0 = s0 + cd_col();
      const int d0 = i - j0;
      const int d1 = d0 - 16;
      const float dv0 = (d0 >= 0) ? exp2f((float)d0 * l2g) : 0.0f;
      const float dv1 = (d1 >= 0) ? exp2f((float)d1 * l2g) : 0.0f;
      tile[m * 32 + cd_col()] = (bf16_t)(s0v[rr] * dv0);
      tile[m * 32 + 16 + cd_col()] = (bf16_t)(s1v[rr] * dv1);
    }
    v16bf aS = load_a_frag_lds(tile, 32);   // compiler inserts s_wait_dscnt
    v16bf bv0 = load_b_frag(vTh + (size_t)0 * 16 * T_ + s0, T_);
    v16bf bv1 = load_b_frag(vTh + (size_t)1 * 16 * T_ + s0, T_);
    v16bf bv2 = load_b_frag(vTh + (size_t)2 * 16 * T_ + s0, T_);
    v16bf bv3 = load_b_frag(vTh + (size_t)3 * 16 * T_ + s0, T_);
    acc0 = wmma_bf16(aS, bv0, acc0);
    acc1 = wmma_bf16(aS, bv1, acc1);
    acc2 = wmma_bf16(aS, bv2, acc2);
    acc3 = wmma_bf16(aS, bv3, acc3);
  }

  // write r = inner + cross, accumulate GroupNorm statistics
  float sum = 0.0f, sq = 0.0f;
  float* rh = rbuf + (size_t)bh * T_ * HD_;
  v8f accs[4] = {acc0, acc1, acc2, acc3};
#pragma unroll
  for (int et = 0; et < 4; ++et) {
    const int e = et * 16 + cd_col();
#pragma unroll
    for (int rr = 0; rr < 8; ++rr) {
      const int t = rowBase + cd_row(rr);
      const float v = accs[et][rr];
      rh[(size_t)t * HD_ + e] = v;
      sum += v;
      sq += v * v;
    }
  }
#pragma unroll
  for (int off = 16; off > 0; off >>= 1) {
    sum += __shfl_xor(sum, off, 32);
    sq += __shfl_xor(sq, off, 32);
  }
  if (lane == 0) {
    atomicAdd(&sums[b], sum);
    atomicAdd(&sums[B_ + b], sq);
  }
}

// ---------------------------------------------------------------------------
// 4) state update GEMM: kv[b,h] = k^T @ v  (HD x HD, K = T), atomic batch sum
//    A double-buffered, 4 B fragments batch-loaded, named accumulators.
// ---------------------------------------------------------------------------
__global__ __launch_bounds__(256) void kv_gemm_kernel(
    const bf16_t* __restrict__ kT, const bf16_t* __restrict__ vT,
    float* __restrict__ kv_accum) {
  const int wave = (int)((blockIdx.x * blockDim.x + threadIdx.x) >> 5); // 0..511
  const int dt = wave & 3;
  const int bh = wave >> 2;
  const int h = bh & (NH_ - 1);
  const bf16_t* kTh = kT + (size_t)bh * HD_ * T_;
  const bf16_t* vTh = vT + (size_t)bh * HD_ * T_;
  const int dBase = dt * 16;

  v8f zero = {};
  v8f c0 = zero, c1 = zero, c2 = zero, c3 = zero;

  const bf16_t* pa = kTh + (size_t)dBase * T_;
  v16bf a = load_a_frag(pa, T_);
  for (int tc = 0; tc < T_ / 32; ++tc) {
    const int t0 = tc * 32;
    v16bf b0 = load_b_frag(vTh + (size_t)0 * 16 * T_ + t0, T_);
    v16bf b1 = load_b_frag(vTh + (size_t)1 * 16 * T_ + t0, T_);
    v16bf b2 = load_b_frag(vTh + (size_t)2 * 16 * T_ + t0, T_);
    v16bf b3 = load_b_frag(vTh + (size_t)3 * 16 * T_ + t0, T_);
    v16bf na = a;
    if (tc + 1 < T_ / 32) na = load_a_frag(pa + t0 + 32, T_);
    c0 = wmma_bf16(a, b0, c0);
    c1 = wmma_bf16(a, b1, c1);
    c2 = wmma_bf16(a, b2, c2);
    c3 = wmma_bf16(a, b3, c3);
    a = na;
  }

  float* kvh = kv_accum + (size_t)h * HD_ * HD_;
  v8f accs[4] = {c0, c1, c2, c3};
#pragma unroll
  for (int et = 0; et < 4; ++et) {
    const int e = et * 16 + cd_col();
#pragma unroll
    for (int rr = 0; rr < 8; ++rr) {
      const int d = dBase + cd_row(rr);
      atomicAdd(&kvh[d * HD_ + e], accs[et][rr]);
    }
  }
}

// current_kv = gamma * past_kv + mean_b(k^T v)
__global__ void kv_final_kernel(const float* __restrict__ past_kv,
                                const float* __restrict__ kv_accum,
                                float* __restrict__ out_kv) {
  int i = blockIdx.x * blockDim.x + threadIdx.x;
  if (i >= NH_ * HD_ * HD_) return;
  int h = i >> 12;
  float gamma = 1.0f - exp2f(-5.0f - (float)h);
  out_kv[i] = gamma * past_kv[i] + kv_accum[i] * (1.0f / (float)B_);
}

// ---------------------------------------------------------------------------
// 5) GroupNorm(1 group) finalize + rearrange to [B, T, C]
// ---------------------------------------------------------------------------
__global__ void gn_final_kernel(const float* __restrict__ rbuf,
                                const float* __restrict__ sums,
                                const float* __restrict__ w,
                                const float* __restrict__ bias,
                                float* __restrict__ out) {
  size_t i = (size_t)blockIdx.x * blockDim.x + threadIdx.x;
  if (i >= (size_t)B_ * T_ * C_) return;
  const int c = (int)(i & (C_ - 1));
  const size_t bt = i >> 10;
  const int t = (int)(bt & (T_ - 1));
  const int b = (int)(bt >> 10);
  const int h = c >> 6;
  const int d = c & 63;
  const float N = (float)C_ * (float)T_;
  const float mu = sums[b] / N;
  const float var = sums[B_ + b] / N - mu * mu;   // biased var (torch GroupNorm)
  const float rs = rsqrtf(var + 1e-5f);
  const float v = rbuf[((size_t)(b * NH_ + h) * T_ + t) * HD_ + d];
  out[i] = (v - mu) * rs * w[h] + bias[h];
}

// ---------------------------------------------------------------------------
// launch
// ---------------------------------------------------------------------------
extern "C" void kernel_launch(void* const* d_in, const int* in_sizes, int n_in,
                              void* d_out, int out_size, void* d_ws, size_t ws_size,
                              hipStream_t stream) {
  (void)in_sizes; (void)n_in; (void)out_size; (void)ws_size;
  const float* x = (const float*)d_in[0];
  const float* past_kv = (const float*)d_in[1];
  const float* W = (const float*)d_in[2];
  const float* gnw = (const float*)d_in[3];
  const float* gnb = (const float*)d_in[4];
  float* out = (float*)d_out;                  // [B,T,C] then [NH,HD,HD]
  float* out_kv = out + (size_t)B_ * T_ * C_;

  // workspace layout (bytes)
  char* ws = (char*)d_ws;
  const size_t SZ_WT = (size_t)N3C_ * C_ * sizeof(bf16_t);        //  6 MB
  const size_t SZ_XB = (size_t)M_ * C_ * sizeof(bf16_t);          // 16 MB
  const size_t SZ_HEAD = (size_t)B_ * NH_ * T_ * HD_ * sizeof(bf16_t); // 16 MB
  const size_t SZ_PKT = (size_t)NH_ * HD_ * HD_ * sizeof(bf16_t);
  const size_t SZ_R = (size_t)B_ * NH_ * T_ * HD_ * sizeof(float); // 32 MB
  const size_t SZ_SUM = 256;                                       // 2*B floats
  size_t off = 0;
  bf16_t* Wt = (bf16_t*)(ws + off); off += SZ_WT;
  bf16_t* xb = (bf16_t*)(ws + off); off += SZ_XB;
  bf16_t* qb = (bf16_t*)(ws + off); off += SZ_HEAD;
  bf16_t* kb = (bf16_t*)(ws + off); off += SZ_HEAD;
  bf16_t* kTb = (bf16_t*)(ws + off); off += SZ_HEAD;
  bf16_t* vTb = (bf16_t*)(ws + off); off += SZ_HEAD;
  bf16_t* pkT = (bf16_t*)(ws + off); off += SZ_PKT;
  float* rbuf = (float*)(ws + off); off += SZ_R;
  float* sums = (float*)(ws + off); off += SZ_SUM;
  float* kv_accum = (float*)(ws + off);

  const int TPB = 256;

  init_kernel<<<(NH_ * HD_ * HD_ + TPB - 1) / TPB, TPB, 0, stream>>>(sums, kv_accum);

  cast_x_kernel<<<(int)(((size_t)M_ * C_ + TPB - 1) / TPB), TPB, 0, stream>>>(x, xb);
  cast_w_kernel<<<(int)(((size_t)N3C_ * C_ + TPB - 1) / TPB), TPB, 0, stream>>>(W, Wt);
  cast_pk_kernel<<<(NH_ * HD_ * HD_ + TPB - 1) / TPB, TPB, 0, stream>>>(past_kv, pkT);

  // QKV GEMM: (M/32)*(3C/64) waves, 8 waves/block
  {
    const int waves = (M_ / 32) * (N3C_ / 64);   // 256*48 = 12288
    qkv_gemm_kernel<<<waves / 8, TPB, 0, stream>>>(xb, Wt, qb, kb, kTb, vTb);
  }

  // retention: B*NH*(T/16) waves, 8 waves/block
  {
    const int waves = B_ * NH_ * (T_ / 16);      // 8192
    retention_kernel<<<waves / 8, TPB, 0, stream>>>(qb, kb, vTb, pkT, rbuf, sums);
  }

  // k^T v state update: B*NH*(HD/16) waves
  {
    const int waves = B_ * NH_ * (HD_ / 16);     // 512
    kv_gemm_kernel<<<waves / 8, TPB, 0, stream>>>(kTb, vTb, kv_accum);
  }
  kv_final_kernel<<<(NH_ * HD_ * HD_ + TPB - 1) / TPB, TPB, 0, stream>>>(
      past_kv, kv_accum, out_kv);

  gn_final_kernel<<<(int)(((size_t)B_ * T_ * C_ + TPB - 1) / TPB), TPB, 0, stream>>>(
      rbuf, sums, gnw, gnb, out);
}
